// PointNet2_FP_Block_68496138437082
// MI455X (gfx1250) — matrix operand order repaired and checked by reading, not compiled, not run on hardware
//
#include <hip/hip_runtime.h>
#include <hip/hip_bf16.h>
#include <stdint.h>

// Problem dims (fixed by reference setup_inputs)
#define BATCH 16
#define NPTS  4096
#define MPTS  1024
#define C1D   256
#define C2D   256
#define CIN   512
#define HID   256

typedef __attribute__((ext_vector_type(16))) __bf16         v16bf;
typedef __attribute__((ext_vector_type(8)))  float          v8f;
typedef __attribute__((ext_vector_type(8)))  unsigned short us8;

__device__ __forceinline__ unsigned short f2bf(float f) {
  union { float f; unsigned u; } v; v.f = f;
  unsigned u = v.u;
  return (unsigned short)((u + 0x7FFFu + ((u >> 16) & 1u)) >> 16); // RNE
}

struct Frag32 { float4 a, b; };
__device__ __forceinline__ v16bf load_frag(const unsigned short* p) {
  Frag32 f;
  f.a = *(const float4*)p;        // ds_load_b128
  f.b = *(const float4*)(p + 16); // ds_load_b128
  return __builtin_bit_cast(v16bf, f);
}

// CDNA5 async global->LDS copy, 16B per lane, tracked by ASYNCcnt.
// LDS byte offset = low 32 bits of the generic pointer to __shared__ memory
__device__ __forceinline__ void async_g2l_b128(unsigned short* lds_ptr, const void* gaddr) {
  unsigned lds_off = (unsigned)(uintptr_t)lds_ptr;
  asm volatile("global_load_async_to_lds_b128 %0, %1, off"
               :: "v"(lds_off), "v"(gaddr)
               : "memory");
}
__device__ __forceinline__ void wait_async0() {
  asm volatile("s_wait_asynccnt 0" ::: "memory");
}

// ---------------------------------------------------------------------------
// Kernel 1: 3-nearest-neighbor search + inverse-distance weights
// grid (N/256, B), block 256. xyz2[b] staged in LDS (12 KB).
// ---------------------------------------------------------------------------
__global__ void knn3_kernel(const float* __restrict__ xyz1,
                            const float* __restrict__ xyz2,
                            int* __restrict__ idx, float* __restrict__ wgt) {
  __shared__ float s2[MPTS * 3];
  const int b  = blockIdx.y;
  const int n  = blockIdx.x * 256 + threadIdx.x;
  const float* src = xyz2 + (size_t)b * MPTS * 3;
  for (int i = threadIdx.x; i < MPTS * 3; i += 256) s2[i] = src[i];
  __syncthreads();

  const float* p = xyz1 + ((size_t)b * NPTS + n) * 3;
  const float px = p[0], py = p[1], pz = p[2];
  float d0 = 3.4e38f, d1 = 3.4e38f, d2 = 3.4e38f;
  int   i0 = 0, i1 = 0, i2 = 0;
#pragma unroll 4
  for (int m = 0; m < MPTS; ++m) {
    const float dx = px - s2[m * 3 + 0];
    const float dy = py - s2[m * 3 + 1];
    const float dz = pz - s2[m * 3 + 2];
    const float d = dx * dx + dy * dy + dz * dz;
    if (d < d0)      { d2 = d1; i2 = i1; d1 = d0; i1 = i0; d0 = d; i0 = m; }
    else if (d < d1) { d2 = d1; i2 = i1; d1 = d;  i1 = m; }
    else if (d < d2) { d2 = d;  i2 = m; }
  }
  float w0 = 1.0f / (d0 + 1e-8f);
  float w1 = 1.0f / (d1 + 1e-8f);
  float w2 = 1.0f / (d2 + 1e-8f);
  const float rs = 1.0f / (w0 + w1 + w2);
  const size_t o = ((size_t)b * NPTS + n) * 3;
  idx[o + 0] = i0; idx[o + 1] = i1; idx[o + 2] = i2;
  wgt[o + 0] = w0 * rs; wgt[o + 1] = w1 * rs; wgt[o + 2] = w2 * rs;
}

// ---------------------------------------------------------------------------
// Kernel 2: convert GEMM weights fp32 -> bf16
// ---------------------------------------------------------------------------
__global__ void cvt_w_kernel(const float* __restrict__ w1, const float* __restrict__ w2,
                             unsigned short* __restrict__ o1, unsigned short* __restrict__ o2) {
  const int i = blockIdx.x * 256 + threadIdx.x;
  if (i < HID * CIN) o1[i] = f2bf(w1[i]);
  if (i < HID * HID) o2[i] = f2bf(w2[i]);
}

// ---------------------------------------------------------------------------
// Kernel 3: build x = concat(interp(feat2), feat1) as bf16 [B][512][N]
// grid (N/256, 8, B), block 256. c-chunks 0..3 = interp rows, 4..7 = copy rows.
// ---------------------------------------------------------------------------
__global__ void build_x_kernel(const float* __restrict__ feat1,
                               const float* __restrict__ feat2,
                               const int* __restrict__ idx,
                               const float* __restrict__ wgt,
                               unsigned short* __restrict__ x) {
  const int b  = blockIdx.z;
  const int n0 = blockIdx.x * 256;
  const int c0 = blockIdx.y * 64;
  const int j  = threadIdx.x;
  const int n  = n0 + j;

  if (c0 < C2D) {
    __shared__ int   si[256 * 3];
    __shared__ float sw[256 * 3];
    const size_t base = ((size_t)b * NPTS + n0) * 3;
    for (int i = j; i < 256 * 3; i += 256) { si[i] = idx[base + i]; sw[i] = wgt[base + i]; }
    __syncthreads();
    const int   i0 = si[j * 3 + 0], i1 = si[j * 3 + 1], i2 = si[j * 3 + 2];
    const float w0 = sw[j * 3 + 0], w1 = sw[j * 3 + 1], w2 = sw[j * 3 + 2];
    const float* f2 = feat2 + (size_t)b * C2D * MPTS;
    unsigned short* xo = x + ((size_t)b * CIN + c0) * NPTS + n;
#pragma unroll 4
    for (int c = c0; c < c0 + 64; ++c) {
      const float* row = f2 + (size_t)c * MPTS;
      *xo = f2bf(w0 * row[i0] + w1 * row[i1] + w2 * row[i2]);
      xo += NPTS;
    }
  } else {
    const float* f1 = feat1 + ((size_t)b * C1D + (c0 - C2D)) * NPTS + n;
    unsigned short* xo = x + ((size_t)b * CIN + c0) * NPTS + n;
#pragma unroll 4
    for (int c = 0; c < 64; ++c) { *xo = f2bf(*f1); f1 += NPTS; xo += NPTS; }
  }
}

// ---------------------------------------------------------------------------
// Kernel 4/5: fused GEMM + BN(eval) + ReLU using v_wmma_f32_16x16x32_bf16
//   out[b][o][n] = relu( (sum_k A[o][k] * X[b][k][n]) * gamma[o]*inv + beta[o] )
// Block = 256 threads (8 waves), 128x128 tile; wave = 4 M-tiles x 2 N-tiles.
// A-tile staged via CDNA5 async global->LDS DMA (no transpose needed).
// B-tile transposed through VGPRs into LDS [n][k] with stride 40 halves.
// ---------------------------------------------------------------------------
#define LSTR 40   // halves; 80 bytes, 16B-aligned, bank-conflict-free row starts

template <int K, bool OUT_BF16>
__global__ __launch_bounds__(256) void gemm_bn_relu_kernel(
    const unsigned short* __restrict__ A,   // [HID][K] bf16
    const unsigned short* __restrict__ Xb,  // [B][K][N] bf16
    const float* __restrict__ gamma, const float* __restrict__ beta,
    void* __restrict__ out) {
  __shared__ __align__(16) unsigned short sA[128 * LSTR];
  __shared__ __align__(16) unsigned short sB[128 * LSTR];

  const int b    = blockIdx.z;
  const int o0   = blockIdx.y * 128;
  const int n0   = blockIdx.x * 128;
  const int tid  = threadIdx.x;
  const int wave = tid >> 5;
  const int lane = tid & 31;
  const int wm   = wave >> 2;   // 0..1 -> 64 rows each
  const int wn   = wave & 3;    // 0..3 -> 32 cols each
  const int l    = lane & 15;
  const int h    = lane >> 4;

  v8f acc[4][2];
#pragma unroll
  for (int t = 0; t < 4; ++t)
#pragma unroll
    for (int u = 0; u < 2; ++u) acc[t][u] = (v8f)(0.0f);

  for (int kk = 0; kk < K; kk += 32) {
    // ---- B tile global reads first (VGPR round-trip; transpose needed) ----
    const int kp  = tid >> 4;              // k pair 0..15 -> k = 2*kp
    const int seg = tid & 15;              // n segment of 8
    const unsigned short* g = Xb + ((size_t)b * K + kk + 2 * kp) * NPTS + n0 + seg * 8;
    us8 e0 = *(const us8*)g;
    us8 e1 = *(const us8*)(g + NPTS);
    if (kk + 32 < K) __builtin_prefetch(g + (size_t)32 * NPTS, 0, 0); // global_prefetch_b8

    // ---- A tile: async DMA global->LDS, 512 chunks of 16B, no transpose ----
#pragma unroll
    for (int c2 = 0; c2 < 2; ++c2) {
      const int cid = tid + c2 * 256;
      const int row = cid >> 2;
      const int ch  = cid & 3;
      async_g2l_b128(&sA[row * LSTR + ch * 8],
                     A + (size_t)(o0 + row) * K + kk + ch * 8);
    }

    // ---- transpose-pack B into LDS: sB[n_local][k_local] ----
#pragma unroll
    for (int jj = 0; jj < 8; ++jj) {
      const int nl = seg * 8 + jj;
      const unsigned v = (unsigned)e0[jj] | ((unsigned)e1[jj] << 16);
      *(unsigned*)&sB[nl * LSTR + 2 * kp] = v;
    }

    wait_async0();
    __syncthreads();

    // ---- load fragments (lanes 0-15: K 0-7/16-23; lanes 16-31: K 8-15/24-31)
    v16bf af[4], bfv[2];
#pragma unroll
    for (int t = 0; t < 4; ++t)
      af[t] = load_frag(&sA[(wm * 64 + t * 16 + l) * LSTR + h * 8]);
#pragma unroll
    for (int u = 0; u < 2; ++u)
      bfv[u] = load_frag(&sB[(wn * 32 + u * 16 + l) * LSTR + h * 8]);

#pragma unroll
    for (int t = 0; t < 4; ++t)
#pragma unroll
      for (int u = 0; u < 2; ++u)
        acc[t][u] = __builtin_amdgcn_wmma_f32_16x16x32_bf16(
            false, af[t], false, bfv[u], (short)0, acc[t][u], false, false);
    __syncthreads();
  }

  // ---- epilogue: BN(eval, running stats identity) + ReLU ----
  const float inv = 0.99999500003749981f;  // 1/sqrt(1 + 1e-5)
#pragma unroll
  for (int t = 0; t < 4; ++t) {
    const int ob = o0 + wm * 64 + t * 16 + h * 8;
    float sc[8], bs[8];
#pragma unroll
    for (int r = 0; r < 8; ++r) { sc[r] = gamma[ob + r] * inv; bs[r] = beta[ob + r]; }
#pragma unroll
    for (int u = 0; u < 2; ++u) {
      const int n = n0 + wn * 32 + u * 16 + l;
#pragma unroll
      for (int r = 0; r < 8; ++r) {
        float v = fmaxf(acc[t][u][r] * sc[r] + bs[r], 0.0f);
        const size_t off = ((size_t)b * HID + ob + r) * NPTS + n;
        if (OUT_BF16) ((unsigned short*)out)[off] = f2bf(v);
        else          ((float*)out)[off] = v;
      }
    }
  }
}

// ---------------------------------------------------------------------------
extern "C" void kernel_launch(void* const* d_in, const int* in_sizes, int n_in,
                              void* d_out, int out_size, void* d_ws, size_t ws_size,
                              hipStream_t stream) {
  const float* xyz1   = (const float*)d_in[0];
  const float* xyz2   = (const float*)d_in[1];
  const float* feat1  = (const float*)d_in[2];
  const float* feat2  = (const float*)d_in[3];
  const float* w1     = (const float*)d_in[4];
  const float* gamma1 = (const float*)d_in[5];
  const float* beta1  = (const float*)d_in[6];
  const float* w2     = (const float*)d_in[7];
  const float* gamma2 = (const float*)d_in[8];
  const float* beta2  = (const float*)d_in[9];

  char* ws = (char*)d_ws;
  int*            idx = (int*)(ws + 0);                         //  786432 B
  float*          wgt = (float*)(ws + 786432);                  //  786432 B
  unsigned short* w1b = (unsigned short*)(ws + 1572864);        //  262144 B
  unsigned short* w2b = (unsigned short*)(ws + 1835008);        //  131072 B
  unsigned short* xb  = (unsigned short*)(ws + 2097152);        // 67108864 B
  unsigned short* y1  = (unsigned short*)(ws + 2097152 + 67108864); // 33554432 B

  knn3_kernel<<<dim3(NPTS / 256, BATCH), 256, 0, stream>>>(xyz1, xyz2, idx, wgt);
  cvt_w_kernel<<<dim3((HID * CIN + 255) / 256), 256, 0, stream>>>(w1, w2, w1b, w2b);
  build_x_kernel<<<dim3(NPTS / 256, CIN / 64, BATCH), 256, 0, stream>>>(feat1, feat2, idx, wgt, xb);
  gemm_bn_relu_kernel<CIN, true><<<dim3(NPTS / 128, HID / 128, BATCH), 256, 0, stream>>>(
      w1b, xb, gamma1, beta1, (void*)y1);
  gemm_bn_relu_kernel<HID, false><<<dim3(NPTS / 128, HID / 128, BATCH), 256, 0, stream>>>(
      w2b, y1, gamma2, beta2, d_out);
}
